// Model_63213328663225
// MI455X (gfx1250) — matrix operand order repaired and checked by reading, not compiled, not run on hardware
//
#include <hip/hip_runtime.h>
#include <hip/hip_bf16.h>
#include <math.h>

#define PI_F 3.14159265358979323846f

typedef __attribute__((ext_vector_type(16))) _Float16 v16h;
typedef __attribute__((ext_vector_type(8)))  float    v8f;
typedef __attribute__((ext_vector_type(4)))  unsigned int v4u;

__device__ __forceinline__ float silu_f(float x) { return x / (1.f + expf(-x)); }
__device__ __forceinline__ int imin(int a, int b) { return a < b ? a : b; }

// ---------------------------------------------------------------------------
// Generic WMMA GEMM:  Y[rows x M] = act( X[rows x K] * W[K x M] + bias ) + Cin
// Block: 256 threads = 8 waves. Block tile 128x64, K step 32.
// Each wave: 16 rows x 64 cols = 4 WMMAs sharing one A fragment.
// Register-staged double-buffered LDS (one barrier per K-step), branch-free
// clamped loads, 16B-aligned contiguous fragments (A stride-40 rows, W
// transposed in LDS). Epilogue specialized by template -> straight-line.
// ---------------------------------------------------------------------------
template<int ACT, bool HAS_BIAS, bool HAS_CIN>
__global__ void __launch_bounds__(256) gemm_wmma_kernel(
    const float* __restrict__ X, int ldx,
    const float* __restrict__ W,          // row-major K x M (ld = M)
    const float* __restrict__ bias,       // length M (if HAS_BIAS)
    const float* __restrict__ Cin,        // rows x M (if HAS_CIN)
    float* __restrict__ Y,                // rows x M (ld = M)
    int rows, int K, int M)
{
    __shared__ _Float16 Xs[2][128][40];   // 80B row stride -> 16B aligned chunks
    __shared__ _Float16 WsT[2][64][40];   // transposed W tile

    const int tid    = threadIdx.x;
    const int lane   = tid & 31;
    const int wid    = tid >> 5;          // wave 0..7 -> rows [wid*16, wid*16+16)
    const int lane16 = lane & 15;
    const int lg     = lane >> 4;
    const int rowBase = blockIdx.y * 128;
    const int colBase = blockIdx.x * 64;

    v8f acc[4];
#pragma unroll
    for (int j = 0; j < 4; ++j) acc[j] = (v8f){0.f,0.f,0.f,0.f,0.f,0.f,0.f,0.f};

    float xstg[8][2];                     // staged X pairs (128x32 tile)
    float wstg[4][2];                     // staged W pairs (32x64 tile)

    auto load_chunk = [&](int k0) {
#pragma unroll
        for (int i = 0; i < 8; ++i) {
            int pi = tid + i * 256;            // pair index 0..2047
            int r  = pi >> 4;
            int kk = (pi & 15) * 2;
            int gr = rowBase + r;
            int grc = imin(gr, rows - 1);
            int g0 = k0 + kk, g1 = g0 + 1;
            const float* xp = X + (size_t)grc * ldx;
            float a0 = xp[imin(g0, K - 1)];
            float a1 = xp[imin(g1, K - 1)];
            xstg[i][0] = (gr < rows && g0 < K) ? a0 : 0.f;
            xstg[i][1] = (gr < rows && g1 < K) ? a1 : 0.f;
        }
#pragma unroll
        for (int i = 0; i < 4; ++i) {
            int pi = tid + i * 256;            // pair index 0..1023
            int c   = pi & 63;
            int kk2 = (pi >> 6) * 2;
            int gc  = colBase + c;
            int gcc = imin(gc, M - 1);
            int gk0 = k0 + kk2, gk1 = gk0 + 1;
            float a0 = W[(size_t)imin(gk0, K - 1) * M + gcc];
            float a1 = W[(size_t)imin(gk1, K - 1) * M + gcc];
            wstg[i][0] = (gc < M && gk0 < K) ? a0 : 0.f;
            wstg[i][1] = (gc < M && gk1 < K) ? a1 : 0.f;
        }
    };

    auto store_chunk = [&](int buf) {
#pragma unroll
        for (int i = 0; i < 8; ++i) {
            int pi = tid + i * 256;
            int r  = pi >> 4;
            int kk = (pi & 15) * 2;
            union { _Float16 h[2]; unsigned int u; } pk;
            pk.h[0] = (_Float16)xstg[i][0]; pk.h[1] = (_Float16)xstg[i][1];
            *(unsigned int*)&Xs[buf][r][kk] = pk.u;
        }
#pragma unroll
        for (int i = 0; i < 4; ++i) {
            int pi = tid + i * 256;
            int c   = pi & 63;
            int kk2 = (pi >> 6) * 2;
            union { _Float16 h[2]; unsigned int u; } pk;
            pk.h[0] = (_Float16)wstg[i][0]; pk.h[1] = (_Float16)wstg[i][1];
            *(unsigned int*)&WsT[buf][c][kk2] = pk.u;
        }
    };

    auto compute_chunk = [&](int buf) {
        union { v16h v; v4u u[2]; } af;
        int rowA = wid * 16 + lane16;
        af.u[0] = *(const v4u*)&Xs[buf][rowA][lg * 8];        // k = lg*8..+7
        af.u[1] = *(const v4u*)&Xs[buf][rowA][lg * 8 + 16];   // k = lg*8+16..+23
#pragma unroll
        for (int j = 0; j < 4; ++j) {
            union { v16h v; v4u u[2]; } bf;
            int rc = j * 16 + lane16;
            bf.u[0] = *(const v4u*)&WsT[buf][rc][lg * 8];      // k = lg*8..+7
            bf.u[1] = *(const v4u*)&WsT[buf][rc][lg * 8 + 16]; // k = lg*8+16..+23
            acc[j] = __builtin_amdgcn_wmma_f32_16x16x32_f16(
                false, af.v, false, bf.v, (short)0, acc[j], false, false);
        }
    };

    const int nk = (K + 31) >> 5;
    load_chunk(0);
    store_chunk(0);
    __syncthreads();
    for (int c = 0; c < nk; ++c) {
        if (c + 1 < nk) load_chunk((c + 1) << 5);   // global -> regs, overlaps WMMA
        compute_chunk(c & 1);
        if (c + 1 < nk) store_chunk((c + 1) & 1);   // regs -> other LDS buffer
        __syncthreads();                            // single barrier per K-step
    }

    // ---- Straight-line epilogue (template-specialized)
#pragma unroll
    for (int j = 0; j < 4; ++j) {
        int col = colBase + j * 16 + lane16;
        float bv = 0.f;
        if (HAS_BIAS && col < M) bv = bias[col];
#pragma unroll
        for (int r = 0; r < 8; ++r) {
            int row = rowBase + wid * 16 + lg * 8 + r;
            if (row < rows && col < M) {
                float v = acc[j][r];
                if (HAS_BIAS) v += bv;
                if (ACT == 1) v = silu_f(v);
                if (HAS_CIN) v += Cin[(size_t)row * M + col];
                Y[(size_t)row * M + col] = v;
            }
        }
    }
}

// ---------------------------------------------------------------------------
// Fourier position encoding (33) + packed [state(4), pos(3)] input rows
// ---------------------------------------------------------------------------
__global__ void encode_kernel(const float* __restrict__ state,
                              const float* __restrict__ pos,
                              float* __restrict__ penc,   // rows x 33
                              float* __restrict__ x7,     // rows x 7
                              int rows)
{
    int row = blockIdx.x * blockDim.x + threadIdx.x;
    if (row >= rows) return;
    const float* p = pos + (size_t)row * 3;
    float* pe = penc + (size_t)row * 33;
#pragma unroll
    for (int d = 0; d < 3; ++d) {
        float x = p[d];
        float f = PI_F;
#pragma unroll
        for (int i = 0; i < 5; ++i) {
            pe[d * 10 + i]     = cosf(x * f);
            pe[d * 10 + 5 + i] = sinf(x * f);
            f *= 2.f;
        }
        pe[30 + d] = x;
    }
    float* xr = x7 + (size_t)row * 7;
    const float* s = state + (size_t)row * 4;
    xr[0] = s[0]; xr[1] = s[1]; xr[2] = s[2]; xr[3] = s[3];
    xr[4] = p[0]; xr[5] = p[1]; xr[6] = p[2];
}

// ---------------------------------------------------------------------------
// Time + condition embedding MLPs -> per-batch broadcast vector (256)
// ---------------------------------------------------------------------------
__global__ void tc_mlp_kernel(const float* __restrict__ time_i,
                              const float* __restrict__ cond,   // b x 32
                              const float* wt0, const float* bt0,
                              const float* wt1, const float* bt1,
                              const float* wc0, const float* bc0,
                              const float* wc1, const float* bc1,
                              float* __restrict__ bcast)        // b x 256
{
    __shared__ float tenc[11], cenc[352], ht[256], hc[256];
    const int b = blockIdx.x, tid = threadIdx.x;
    float t = time_i[b];
    if (tid < 11) {
        if (tid < 5)       tenc[tid] = cosf(t * PI_F * (float)(1 << tid));
        else if (tid < 10) tenc[tid] = sinf(t * PI_F * (float)(1 << (tid - 5)));
        else               tenc[10]  = t;
    }
    for (int j = tid; j < 352; j += 256) {
        if (j < 320) {
            int d = j / 10, i = j % 10;
            float x = cond[b * 32 + d];
            cenc[j] = (i < 5) ? cosf(x * PI_F * (float)(1 << i))
                              : sinf(x * PI_F * (float)(1 << (i - 5)));
        } else {
            cenc[j] = cond[b * 32 + (j - 320)];
        }
    }
    __syncthreads();
    {
        float a = bt0[tid];
        for (int i = 0; i < 11; ++i) a += tenc[i] * wt0[i * 256 + tid];
        ht[tid] = silu_f(a);
        float c = bc0[tid];
        for (int i = 0; i < 352; ++i) c += cenc[i] * wc0[i * 256 + tid];
        hc[tid] = silu_f(c);
    }
    __syncthreads();
    float o = bt1[tid] + bc1[tid];
    for (int i = 0; i < 256; ++i)
        o += ht[i] * wt1[i * 256 + tid] + hc[i] * wc1[i * 256 + tid];
    bcast[b * 256 + tid] = o;
}

__global__ void add_bcast_kernel(float* __restrict__ V,
                                 const float* __restrict__ bcast,
                                 int perBatch, int total)
{
    int idx = blockIdx.x * blockDim.x + threadIdx.x;
    if (idx >= total) return;
    V[idx] += bcast[(idx / perBatch) * 256 + (idx & 255)];
}

__global__ void concat_kernel(float* __restrict__ x,           // rows x 289
                              const float* __restrict__ V,     // rows x 256
                              const float* __restrict__ penc,  // rows x 33
                              int total)
{
    int idx = blockIdx.x * blockDim.x + threadIdx.x;
    if (idx >= total) return;
    int row = idx / 289, j = idx % 289;
    x[idx] = (j < 256) ? V[(size_t)row * 256 + j]
                       : penc[(size_t)row * 33 + (j - 256)];
}

// Wave-per-row layernorm, wave32 shuffles
__global__ void ln_kernel(const float* __restrict__ X,
                          const float* __restrict__ g,
                          const float* __restrict__ bt,
                          float* __restrict__ Y, int rows, int D)
{
    int warp = threadIdx.x >> 5, lane = threadIdx.x & 31;
    int row = blockIdx.x * 8 + warp;
    if (row >= rows) return;
    const float* x = X + (size_t)row * D;
    float s = 0.f;
    for (int j = lane; j < D; j += 32) s += x[j];
    for (int o = 16; o > 0; o >>= 1) s += __shfl_xor(s, o);
    float mu = s / (float)D;
    float v = 0.f;
    for (int j = lane; j < D; j += 32) { float d = x[j] - mu; v += d * d; }
    for (int o = 16; o > 0; o >>= 1) v += __shfl_xor(v, o);
    float rs = rsqrtf(v / (float)D + 1e-5f);
    float* y = Y + (size_t)row * D;
    for (int j = lane; j < D; j += 32) y[j] = (x[j] - mu) * rs * g[j] + bt[j];
}

// In-place softmax over 64 slice logits, one thread per row
__global__ void softmax64_kernel(float* __restrict__ W, int rows)
{
    int row = blockIdx.x * blockDim.x + threadIdx.x;
    if (row >= rows) return;
    float* p = W + (size_t)row * 64;
    float mx = -1e30f;
    for (int m = 0; m < 64; ++m) mx = fmaxf(mx, p[m]);
    float s = 0.f;
    for (int m = 0; m < 64; ++m) { float e = expf(p[m] - mx); p[m] = e; s += e; }
    float inv = 1.f / s;
    for (int m = 0; m < 64; ++m) p[m] *= inv;
}

// wT[(b*64+m)*Nper + n] = W[(b*Nper+n)*64 + m]
__global__ void wtrans_kernel(const float* __restrict__ W,
                              float* __restrict__ wT, int Nper, int total)
{
    int idx = blockIdx.x * blockDim.x + threadIdx.x;
    if (idx >= total) return;
    int n = idx % Nper, bm = idx / Nper;
    int m = bm & 63, b = bm >> 6;
    wT[idx] = W[((size_t)(b * Nper + n)) * 64 + m];
}

// Deterministic row sums of wT (128 rows of length Nper)
__global__ void rowsum_kernel(const float* __restrict__ wT,
                              float* __restrict__ colsum, int Nper)
{
    __shared__ float sm[256];
    int r = blockIdx.x;
    const float* p = wT + (size_t)r * Nper;
    float a = 0.f;
    for (int n = threadIdx.x; n < Nper; n += 256) a += p[n];
    sm[threadIdx.x] = a;
    __syncthreads();
    for (int s = 128; s > 0; s >>= 1) {
        if (threadIdx.x < s) sm[threadIdx.x] += sm[threadIdx.x + s];
        __syncthreads();
    }
    if (threadIdx.x == 0) colsum[r] = sm[0];
}

__global__ void wscale_kernel(float* __restrict__ wT,
                              const float* __restrict__ colsum,
                              int Nper, int total)
{
    int idx = blockIdx.x * blockDim.x + threadIdx.x;
    if (idx >= total) return;
    wT[idx] /= (colsum[idx / Nper] + 1e-6f);
}

// Online-softmax attention: thread per (row, head). H=17, HD=17, M=64.
__global__ void attn_kernel(const float* __restrict__ Q,     // rows x 289
                            const float* __restrict__ KV,    // b x 64 x 578
                            float* __restrict__ O,           // rows x 289
                            int rowsTotal, int Nper)
{
    int idx = blockIdx.x * blockDim.x + threadIdx.x;
    if (idx >= rowsTotal * 17) return;
    int row = idx / 17, h = idx % 17;
    int b = row / Nper;
    const float* q  = Q + (size_t)row * 289 + h * 17;
    const float* kb = KV + (size_t)b * 64 * 578 + h * 17;
    const float* vb = kb + 289;
    float qr[17];
#pragma unroll
    for (int d = 0; d < 17; ++d) qr[d] = q[d];
    float m0 = -1e30f, l = 0.f, o[17];
#pragma unroll
    for (int d = 0; d < 17; ++d) o[d] = 0.f;
    const float scale = 0.2425356250363330f;   // 1/sqrt(17)
    for (int m = 0; m < 64; ++m) {
        const float* kk = kb + m * 578;
        float s = 0.f;
#pragma unroll
        for (int d = 0; d < 17; ++d) s += qr[d] * kk[d];
        s *= scale;
        float nm   = fmaxf(m0, s);
        float corr = expf(m0 - nm);
        float p    = expf(s - nm);
        l = l * corr + p;
        const float* vv = vb + m * 578;
#pragma unroll
        for (int d = 0; d < 17; ++d) o[d] = o[d] * corr + p * vv[d];
        m0 = nm;
    }
    float inv = 1.f / l;
    float* op = O + (size_t)row * 289 + h * 17;
#pragma unroll
    for (int d = 0; d < 17; ++d) op[d] = o[d] * inv;
}

__global__ void silu_kernel(float* __restrict__ p, int n)
{
    int idx = blockIdx.x * blockDim.x + threadIdx.x;
    if (idx < n) p[idx] = silu_f(p[idx]);
}

__global__ void final_kernel(const float* __restrict__ state,
                             const float* __restrict__ vpred,
                             float* __restrict__ out, int n)
{
    int idx = blockIdx.x * blockDim.x + threadIdx.x;
    if (idx < n) out[idx] = state[idx] + 0.05f * vpred[idx];
}

// ---------------------------------------------------------------------------
// Host-side orchestration
// ---------------------------------------------------------------------------
static inline int cdiv(int a, int b) { return (a + b - 1) / b; }

static void gemm(hipStream_t s, const float* X, int ldx, const float* W,
                 const float* bias, const float* Cin, float* Y,
                 int rows, int K, int M, int act)
{
    dim3 grid(cdiv(M, 64), cdiv(rows, 128));
#define GEMM_L(A, HB, HC) \
    gemm_wmma_kernel<A, HB, HC><<<grid, 256, 0, s>>>(X, ldx, W, bias, Cin, Y, rows, K, M)
    if (act == 1) {
        if (bias) { if (Cin) GEMM_L(1, true, true);  else GEMM_L(1, true, false); }
        else      { if (Cin) GEMM_L(1, false, true); else GEMM_L(1, false, false); }
    } else {
        if (bias) { if (Cin) GEMM_L(0, true, true);  else GEMM_L(0, true, false); }
        else      { if (Cin) GEMM_L(0, false, true); else GEMM_L(0, false, false); }
    }
#undef GEMM_L
}

extern "C" void kernel_launch(void* const* d_in, const int* in_sizes, int n_in,
                              void* d_out, int out_size, void* d_ws, size_t ws_size,
                              hipStream_t stream)
{
    (void)in_sizes; (void)n_in; (void)out_size; (void)ws_size;
    const int B = 2, N = 32768, R = B * N;   // rows total

    const float* state_in = (const float*)d_in[0];
    const float* node_pos = (const float*)d_in[1];
    /* d_in[2] = edges (ignored by reference) */
    const float* time_i   = (const float*)d_in[3];
    const float* cond     = (const float*)d_in[4];

    int pi = 5;
    auto nx = [&]() { return (const float*)d_in[pi++]; };
    const float *fv1_w0 = nx(), *fv1_b0 = nx(), *fv1_w1 = nx(), *fv1_b1 = nx();
    const float *fvt_w0 = nx(), *fvt_b0 = nx(), *fvt_w1 = nx(), *fvt_b1 = nx();
    const float *fvc_w0 = nx(), *fvc_b0 = nx(), *fvc_w1 = nx(), *fvc_b1 = nx();
    const float *dec0_w = nx(), *dec0_b = nx();
    const float *dec1_w = nx(), *dec1_b = nx();
    const float *dec2_w = nx(), *dec2_b = nx();
    struct Blk {
        const float *ln1_g, *ln1_b, *ln2_g, *ln2_b;
        const float *slc_w, *slc_b, *kv_w, *q_w, *op_w, *op_b;
        const float *f0_w, *f0_b, *f1_w, *f1_b, *po_w, *po_b;
    } blk[4];
    for (int i = 0; i < 4; ++i) {
        blk[i].ln1_g = nx(); blk[i].ln1_b = nx();
        blk[i].ln2_g = nx(); blk[i].ln2_b = nx();
        blk[i].slc_w = nx(); blk[i].slc_b = nx();
        blk[i].kv_w  = nx(); blk[i].q_w   = nx();
        blk[i].op_w  = nx(); blk[i].op_b  = nx();
        blk[i].f0_w  = nx(); blk[i].f0_b  = nx();
        blk[i].f1_w  = nx(); blk[i].f1_b  = nx();
        blk[i].po_w  = nx(); blk[i].po_b  = nx();
    }

    // Workspace carve-up (floats)
    float* ws = (float*)d_ws;
    size_t off = 0;
    auto alloc = [&](size_t n) { float* p = ws + off; off += n; return p; };
    float* POSENC = alloc((size_t)R * 33);
    float* X7     = alloc((size_t)R * 7);
    float* VENC   = alloc((size_t)R * 256);
    float* VALL   = alloc((size_t)4 * R * 256);
    float* Xb     = alloc((size_t)R * 289);   // V_in (residual stream)
    float* XN     = alloc((size_t)R * 289);   // ln out / attn out / dec hidden
    float* Qb     = alloc((size_t)R * 289);   // q / dec hidden 2
    float* H2     = alloc((size_t)R * 578);   // ffn hidden
    float* Wb     = alloc((size_t)R * 64);    // slice weights
    float* WT     = alloc((size_t)B * 64 * N);
    float* CS     = alloc(128);               // column sums
    float* TOK    = alloc((size_t)B * 64 * 289);
    float* KVB    = alloc((size_t)B * 64 * 578);
    float* BC     = alloc(512);               // time+cond broadcast
    float* Pb     = alloc((size_t)R * 4);     // v_pred

    // --- Encoder ---
    encode_kernel<<<cdiv(R, 256), 256, 0, stream>>>(state_in, node_pos, POSENC, X7, R);
    tc_mlp_kernel<<<B, 256, 0, stream>>>(time_i, cond, fvt_w0, fvt_b0, fvt_w1, fvt_b1,
                                         fvc_w0, fvc_b0, fvc_w1, fvc_b1, BC);
    gemm(stream, X7, 7, fv1_w0, fv1_b0, nullptr, XN, R, 7, 256, /*silu*/1);
    gemm(stream, XN, 256, fv1_w1, fv1_b1, nullptr, VENC, R, 256, 256, 0);
    add_bcast_kernel<<<cdiv(R * 256, 256), 256, 0, stream>>>(VENC, BC, N * 256, R * 256);

    // --- Transolver blocks ---
    const float* Vprev = VENC;
    for (int i = 0; i < 4; ++i) {
        const Blk& p = blk[i];
        concat_kernel<<<cdiv(R * 289, 256), 256, 0, stream>>>(Xb, Vprev, POSENC, R * 289);
        ln_kernel<<<cdiv(R, 8), 256, 0, stream>>>(Xb, p.ln1_g, p.ln1_b, XN, R, 289);

        // slice weights: softmax(xn @ slice_w + b), column-normalize
        gemm(stream, XN, 289, p.slc_w, p.slc_b, nullptr, Wb, R, 289, 64, 0);
        softmax64_kernel<<<cdiv(R, 256), 256, 0, stream>>>(Wb, R);
        wtrans_kernel<<<cdiv(B * 64 * N, 256), 256, 0, stream>>>(Wb, WT, N, B * 64 * N);
        rowsum_kernel<<<B * 64, 256, 0, stream>>>(WT, CS, N);
        wscale_kernel<<<cdiv(B * 64 * N, 256), 256, 0, stream>>>(WT, CS, N, B * 64 * N);

        // tokens (64 x 289 per batch, K = N) and kv (64 x 578 per batch)
        for (int b = 0; b < B; ++b) {
            gemm(stream, WT + (size_t)b * 64 * N, N, XN + (size_t)b * N * 289,
                 nullptr, nullptr, TOK + (size_t)b * 64 * 289, 64, N, 289, 0);
            gemm(stream, TOK + (size_t)b * 64 * 289, 289, p.kv_w,
                 nullptr, nullptr, KVB + (size_t)b * 64 * 578, 64, 289, 578, 0);
        }

        // q, attention, out_proj (+residual into Xb)
        gemm(stream, XN, 289, p.q_w, nullptr, nullptr, Qb, R, 289, 289, 0);
        attn_kernel<<<cdiv(R * 17, 256), 256, 0, stream>>>(Qb, KVB, XN, R, N);
        gemm(stream, XN, 289, p.op_w, p.op_b, Xb, Xb, R, 289, 289, 0);

        // FFN with residual
        ln_kernel<<<cdiv(R, 8), 256, 0, stream>>>(Xb, p.ln2_g, p.ln2_b, XN, R, 289);
        gemm(stream, XN, 289, p.f0_w, p.f0_b, nullptr, H2, R, 289, 578, /*silu*/1);
        gemm(stream, H2, 578, p.f1_w, p.f1_b, Xb, Xb, R, 578, 289, 0);

        // V = V_prev + V_in @ proj_out + b
        float* Vout = VALL + (size_t)i * R * 256;
        gemm(stream, Xb, 289, p.po_w, p.po_b, Vprev, Vout, R, 289, 256, 0);
        Vprev = Vout;
    }

    // --- Decoder: dec0 over concat(V1..V4, pos_enc) as 5 accumulating GEMMs ---
    gemm(stream, VALL, 256, dec0_w, dec0_b, nullptr, XN, R, 256, 256, 0);
    for (int i = 1; i < 4; ++i)
        gemm(stream, VALL + (size_t)i * R * 256, 256, dec0_w + (size_t)i * 256 * 256,
             nullptr, XN, XN, R, 256, 256, 0);
    gemm(stream, POSENC, 33, dec0_w + (size_t)1024 * 256, nullptr, XN, XN, R, 33, 256, 0);
    silu_kernel<<<cdiv(R * 256, 256), 256, 0, stream>>>(XN, R * 256);

    gemm(stream, XN, 256, dec1_w, dec1_b, nullptr, Qb, R, 256, 256, /*silu*/1);
    gemm(stream, Qb, 256, dec2_w, dec2_b, nullptr, Pb, R, 256, 4, 0);
    final_kernel<<<cdiv(R * 4, 256), 256, 0, stream>>>(state_in, Pb, (float*)d_out, R * 4);
}
